// EdgeAwareFloodModel_54382875902415
// MI455X (gfx1250) — matrix operand order repaired and checked by reading, not compiled, not run on hardware
//
#include <hip/hip_runtime.h>
#include <stdint.h>

#define HD 128

typedef __attribute__((ext_vector_type(16))) __bf16 v16bf;
typedef __attribute__((ext_vector_type(8)))  float  v8f;

union Frag16 { v16bf v; uint32_t u[8]; };

__device__ __forceinline__ uint16_t f2bf(float f) {
    uint32_t u = __builtin_bit_cast(uint32_t, f);
    uint32_t r = u + 0x7FFFu + ((u >> 16) & 1u);
    return (uint16_t)(r >> 16);
}
__device__ __forceinline__ float geluf(float x) {
    return 0.5f * x * (1.0f + erff(x * 0.70710678118654752f));
}
__device__ __forceinline__ float sigmf(float x) { return 1.0f / (1.0f + __expf(-x)); }

__device__ __forceinline__ float wsum(float v) {
#pragma unroll
    for (int o = 16; o > 0; o >>= 1) v += __shfl_xor(v, o, 32);
    return v;
}
__device__ __forceinline__ v8f vzero() {
    v8f z;
#pragma unroll
    for (int i = 0; i < 8; ++i) z[i] = 0.0f;
    return z;
}

// A fragment (16x32 bf16, MxK): lane L holds row m0+(L%16); VGPR0-3 K=half*8+0..7, VGPR4-7 K=16+half*8+0..7
__device__ __forceinline__ void loadA(Frag16& a, const uint16_t* row, int k0, int half) {
    const uint32_t* p = (const uint32_t*)row;
    int b0 = (k0 + half * 8) >> 1;
    int b1 = (k0 + 16 + half * 8) >> 1;
    a.u[0] = p[b0]; a.u[1] = p[b0 + 1]; a.u[2] = p[b0 + 2]; a.u[3] = p[b0 + 3];
    a.u[4] = p[b1]; a.u[5] = p[b1 + 1]; a.u[6] = p[b1 + 2]; a.u[7] = p[b1 + 3];
}
// B fragment (32x16 bf16, KxN): lane L holds col n0+(L%16); VGPR0-3 K=half*8+0..7, VGPR4-7 K=16+half*8+0..7
__device__ __forceinline__ void loadB(Frag16& b, const uint16_t* wrow, int k0, int half) {
    const uint32_t* p = (const uint32_t*)wrow;
    int base = (k0 + half * 8) >> 1;
#pragma unroll
    for (int i = 0; i < 8; ++i) b.u[i] = p[base + (i < 4 ? i : i + 4)];
}

#define WMMA_BF16(A, B, C) \
    __builtin_amdgcn_wmma_f32_16x16x32_bf16(false, (A), false, (B), (short)0, (C), false, false)

// acc[8] covers 16 rows x 128 cols.  out = x @ w.T :  B[k][n] = w[n][k], w row-major (O x Kw)
template <int KSTEPS>
__device__ __forceinline__ void gemm(v8f acc[8], const uint16_t* arow, const uint16_t* w,
                                     int wRow0, int Kw, int kOffB, int half, int l16) {
#pragma unroll
    for (int ks = 0; ks < KSTEPS; ++ks) {
        int k0 = ks * 32;
        Frag16 a; loadA(a, arow, k0, half);
#pragma unroll
        for (int nt = 0; nt < 8; ++nt) {
            Frag16 b; loadB(b, w + (size_t)(wRow0 + nt * 16 + l16) * Kw, kOffB + k0, half);
            acc[nt] = WMMA_BF16(a.v, b.v, acc[nt]);
        }
    }
}

// ---------------- weight f32 -> bf16 pre-conversion ----------------
struct WConv { const float* src; unsigned long long dstOff; unsigned cnt; };
struct WTab { WConv w[14]; };

__global__ void k_conv(WTab t, uint16_t* __restrict__ dst) {
    unsigned stride = gridDim.x * blockDim.x;
    for (int i = 0; i < 14; ++i) {
        const float* s = t.w[i].src;
        uint16_t* d = dst + t.w[i].dstOff;
        for (unsigned j = blockIdx.x * blockDim.x + threadIdx.x; j < t.w[i].cnt; j += stride)
            d[j] = f2bf(s[j]);
    }
}

// ---------------- encoders: lin(12/9 -> 128) + LN + GELU + residual ----------------
__global__ __launch_bounds__(256) void k_encode_nodes(
    const float* __restrict__ dyn, const float* __restrict__ stat, const float* __restrict__ st,
    const float* __restrict__ w, const float* __restrict__ b,
    const float* __restrict__ g, const float* __restrict__ beta,
    float* __restrict__ nb32, uint16_t* __restrict__ nb16, uint16_t* __restrict__ ns16,
    float* __restrict__ agg, float* __restrict__ agg2, float* __restrict__ indeg, int N) {
    int gw = (blockIdx.x * blockDim.x + threadIdx.x) >> 5;
    int lane = threadIdx.x & 31;
    if (gw >= N) return;
    size_t n = (size_t)gw;
    float x[12];
#pragma unroll
    for (int i = 0; i < 4; ++i) x[i] = dyn[n * 4 + i];
#pragma unroll
    for (int i = 0; i < 8; ++i) x[4 + i] = stat[n * 8 + i];
    float y[4];
#pragma unroll
    for (int j = 0; j < 4; ++j) {
        int o = lane + 32 * j;
        const float* wr = w + o * 12;
        float s = b[o];
#pragma unroll
        for (int i = 0; i < 12; ++i) s += x[i] * wr[i];
        y[j] = s;
    }
    float m = wsum(y[0] + y[1] + y[2] + y[3]) * (1.0f / 128.0f);
    float vv = 0.f;
#pragma unroll
    for (int j = 0; j < 4; ++j) { float d = y[j] - m; vv += d * d; }
    float inv = rsqrtf(wsum(vv) * (1.0f / 128.0f) + 1e-5f);
#pragma unroll
    for (int j = 0; j < 4; ++j) {
        int o = lane + 32 * j;
        float t = geluf((y[j] - m) * inv * g[o] + beta[o]);
        float s0 = st[n * HD + o];
        float base = t + s0;
        nb32[n * HD + o] = base;
        nb16[n * HD + o] = f2bf(base);
        ns16[n * HD + o] = f2bf(s0);
        agg[n * HD + o] = 0.f;
        agg2[n * HD + o] = 0.f;
    }
    if (lane == 0) indeg[n] = 0.f;
}

__global__ __launch_bounds__(256) void k_encode_edges(
    const float* __restrict__ dyn, const float* __restrict__ stat, const float* __restrict__ st,
    const float* __restrict__ w, const float* __restrict__ b,
    const float* __restrict__ g, const float* __restrict__ beta,
    const int* __restrict__ dst,
    uint16_t* __restrict__ eb16, uint16_t* __restrict__ es16,
    float* __restrict__ indeg, int E) {
    int gw = (blockIdx.x * blockDim.x + threadIdx.x) >> 5;
    int lane = threadIdx.x & 31;
    if (gw >= E) return;
    size_t e = (size_t)gw;
    float x[9];
#pragma unroll
    for (int i = 0; i < 3; ++i) x[i] = dyn[e * 3 + i];
#pragma unroll
    for (int i = 0; i < 6; ++i) x[3 + i] = stat[e * 6 + i];
    float y[4];
#pragma unroll
    for (int j = 0; j < 4; ++j) {
        int o = lane + 32 * j;
        const float* wr = w + o * 9;
        float s = b[o];
#pragma unroll
        for (int i = 0; i < 9; ++i) s += x[i] * wr[i];
        y[j] = s;
    }
    float m = wsum(y[0] + y[1] + y[2] + y[3]) * (1.0f / 128.0f);
    float vv = 0.f;
#pragma unroll
    for (int j = 0; j < 4; ++j) { float d = y[j] - m; vv += d * d; }
    float inv = rsqrtf(wsum(vv) * (1.0f / 128.0f) + 1e-5f);
#pragma unroll
    for (int j = 0; j < 4; ++j) {
        int o = lane + 32 * j;
        float t = geluf((y[j] - m) * inv * g[o] + beta[o]);
        float s0 = st[e * HD + o];
        eb16[e * HD + o] = f2bf(t + s0);
        es16[e * HD + o] = f2bf(s0);
    }
    if (lane == 0) atomicAdd(&indeg[dst[e]], 1.0f);
}

// ---------------- edge message MLP (gather -> 384->128 -> GELU -> 128->128 -> scatter-add) ----------------
template <int HOP>
__global__ __launch_bounds__(256) void k_edge_msg(
    const uint16_t* __restrict__ nsrc16, const uint16_t* __restrict__ eb16,
    const int* __restrict__ src, const int* __restrict__ dst,
    const uint16_t* __restrict__ wa, const float* __restrict__ ba,
    const uint16_t* __restrict__ wb, const float* __restrict__ bb,
    float* __restrict__ agg, uint16_t* __restrict__ msg16, int E) {
    __shared__ uint16_t hid[8][16 * HD];
    int wv = threadIdx.x >> 5, lane = threadIdx.x & 31;
    int half = lane >> 4, l16 = lane & 15;
    int e0 = (blockIdx.x * 8 + wv) * 16;
    int eM = e0 + l16; if (eM >= E) eM = E - 1;
    const uint16_t* pS = nsrc16 + (size_t)src[eM] * HD;
    const uint16_t* pD = nsrc16 + (size_t)dst[eM] * HD;
    const uint16_t* pE = eb16 + (size_t)eM * HD;

    v8f acc[8];
#pragma unroll
    for (int i = 0; i < 8; ++i) acc[i] = vzero();
    gemm<4>(acc, pS, wa, 0, 384, 0, half, l16);
    gemm<4>(acc, pD, wa, 0, 384, 128, half, l16);
    gemm<4>(acc, pE, wa, 0, 384, 256, half, l16);

    uint16_t* hrow = hid[wv];
#pragma unroll
    for (int nt = 0; nt < 8; ++nt) {
        int col = nt * 16 + l16;
        float bias = ba[col];
#pragma unroll
        for (int v = 0; v < 8; ++v) {
            int row = v + 8 * half;
            hrow[row * HD + col] = f2bf(geluf(acc[nt][v] + bias));
        }
    }
    __syncthreads();

    v8f acc2[8];
#pragma unroll
    for (int i = 0; i < 8; ++i) acc2[i] = vzero();
    gemm<4>(acc2, hrow + (size_t)l16 * HD, wb, 0, HD, 0, half, l16);

    int drow[16];
#pragma unroll
    for (int r = 0; r < 16; ++r) { int e = e0 + r; drow[r] = dst[e < E ? e : (E - 1)]; }
#pragma unroll
    for (int nt = 0; nt < 8; ++nt) {
        int col = nt * 16 + l16;
        float bias = bb[col];
#pragma unroll
        for (int v = 0; v < 8; ++v) {
            int row = v + 8 * half;
            int e = e0 + row;
            if (e < E) {
                float val = acc2[nt][v] + bias;
                if (HOP == 1) msg16[(size_t)e * HD + col] = f2bf(val);
                atomicAdd(&agg[(size_t)drow[row] * HD + col], val);
            }
        }
    }
}

// ---------------- node hop1: node_h1 = node_base + GELU(mn1 @ (agg/deg)) ----------------
__global__ __launch_bounds__(256) void k_node_hop1(
    const float* __restrict__ agg, const float* __restrict__ indeg,
    const uint16_t* __restrict__ w, const float* __restrict__ b,
    const float* __restrict__ nb32, uint16_t* __restrict__ nh116, int N) {
    __shared__ uint16_t at[8][16 * HD];
    int wv = threadIdx.x >> 5, lane = threadIdx.x & 31;
    int half = lane >> 4, l16 = lane & 15;
    int n0 = (blockIdx.x * 8 + wv) * 16;
    for (int r = 0; r < 16; ++r) {
        int n = n0 + r;
#pragma unroll
        for (int j = 0; j < 4; ++j) {
            int c = lane + 32 * j;
            float val = 0.f;
            if (n < N) {
                float dg = indeg[n]; if (dg < 1.f) dg = 1.f;
                val = agg[(size_t)n * HD + c] / dg;
            }
            at[wv][r * HD + c] = f2bf(val);
        }
    }
    __syncthreads();
    v8f acc[8];
#pragma unroll
    for (int i = 0; i < 8; ++i) acc[i] = vzero();
    gemm<4>(acc, at[wv] + (size_t)l16 * HD, w, 0, HD, 0, half, l16);
#pragma unroll
    for (int nt = 0; nt < 8; ++nt) {
        int col = nt * 16 + l16;
        float bias = b[col];
#pragma unroll
        for (int v = 0; v < 8; ++v) {
            int row = v + 8 * half;
            int n = n0 + row;
            if (n < N) {
                float val = nb32[(size_t)n * HD + col] + geluf(acc[nt][v] + bias);
                nh116[(size_t)n * HD + col] = f2bf(val);
            }
        }
    }
}

// ---------------- node update: mn2 -> nu -> GRU(g1/g2) -> LN ----------------
__global__ __launch_bounds__(128) void k_node_update(
    const float* __restrict__ agg2, const float* __restrict__ indeg,
    const uint16_t* __restrict__ wmn2, const float* __restrict__ bmn2,
    const uint16_t* __restrict__ nh116, const uint16_t* __restrict__ wnu, const float* __restrict__ bnu,
    const uint16_t* __restrict__ g1ih, const uint16_t* __restrict__ g1hh,
    const float* __restrict__ g1bih, const float* __restrict__ g1bhh,
    const uint16_t* __restrict__ g2ih, const uint16_t* __restrict__ g2hh,
    const float* __restrict__ g2bih, const float* __restrict__ g2bhh,
    const float* __restrict__ hstate, const uint16_t* __restrict__ hs16,
    const float* __restrict__ lng, const float* __restrict__ lnb,
    float* __restrict__ out, int N, int NUM1D) {
    __shared__ uint16_t P[4][16 * HD];
    __shared__ uint16_t Q[4][16 * HD];
    __shared__ float S1[4][16 * HD];
    __shared__ float S2[4][16 * HD];
    int wv = threadIdx.x >> 5, lane = threadIdx.x & 31;
    int half = lane >> 4, l16 = lane & 15;
    int n0 = (blockIdx.x * 4 + wv) * 16;

    for (int r = 0; r < 16; ++r) {
        int n = n0 + r;
#pragma unroll
        for (int j = 0; j < 4; ++j) {
            int c = lane + 32 * j;
            float val = 0.f;
            if (n < N) {
                float dg = indeg[n]; if (dg < 1.f) dg = 1.f;
                val = agg2[(size_t)n * HD + c] / dg;
            }
            P[wv][r * HD + c] = f2bf(val);
        }
    }
    v8f acc[8];
#pragma unroll
    for (int i = 0; i < 8; ++i) acc[i] = vzero();
    gemm<4>(acc, P[wv] + (size_t)l16 * HD, wmn2, 0, HD, 0, half, l16);
#pragma unroll
    for (int nt = 0; nt < 8; ++nt) {
        int col = nt * 16 + l16;
        float bias = bmn2[col];
#pragma unroll
        for (int v = 0; v < 8; ++v) {
            int row = v + 8 * half;
            P[wv][row * HD + col] = f2bf(geluf(acc[nt][v] + bias));
        }
    }

    int nM = n0 + l16; if (nM >= N) nM = N - 1;
#pragma unroll
    for (int i = 0; i < 8; ++i) acc[i] = vzero();
    gemm<4>(acc, nh116 + (size_t)nM * HD, wnu, 0, 256, 0, half, l16);
    gemm<4>(acc, P[wv] + (size_t)l16 * HD, wnu, 0, 256, 128, half, l16);
#pragma unroll
    for (int nt = 0; nt < 8; ++nt) {
        int col = nt * 16 + l16;
        float bias = bnu[col];
#pragma unroll
        for (int v = 0; v < 8; ++v) {
            int row = v + 8 * half;
            Q[wv][row * HD + col] = f2bf(geluf(acc[nt][v] + bias));
        }
    }

    bool is1 = (n0 < NUM1D);
    const uint16_t* wih = is1 ? g1ih : g2ih;
    const uint16_t* whh = is1 ? g1hh : g2hh;
    const float* bih = is1 ? g1bih : g2bih;
    const float* bhh = is1 ? g1bhh : g2bhh;
    const uint16_t* hrow = hs16 + (size_t)nM * HD;
    const uint16_t* xrow = Q[wv] + (size_t)l16 * HD;

    // r gate
#pragma unroll
    for (int i = 0; i < 8; ++i) acc[i] = vzero();
    gemm<4>(acc, xrow, wih, 0, HD, 0, half, l16);
    gemm<4>(acc, hrow, whh, 0, HD, 0, half, l16);
#pragma unroll
    for (int nt = 0; nt < 8; ++nt) {
        int col = nt * 16 + l16;
#pragma unroll
        for (int v = 0; v < 8; ++v) {
            int row = v + 8 * half;
            S1[wv][row * HD + col] = sigmf(acc[nt][v] + bih[col] + bhh[col]);
        }
    }
    // n gate: hidden part scaled by r
#pragma unroll
    for (int i = 0; i < 8; ++i) acc[i] = vzero();
    gemm<4>(acc, hrow, whh, 256, HD, 0, half, l16);
#pragma unroll
    for (int nt = 0; nt < 8; ++nt) {
        int col = nt * 16 + l16;
#pragma unroll
        for (int v = 0; v < 8; ++v) {
            int row = v + 8 * half;
            S2[wv][row * HD + col] = S1[wv][row * HD + col] * (acc[nt][v] + bhh[256 + col]);
        }
    }
    // n gate: input part
#pragma unroll
    for (int i = 0; i < 8; ++i) acc[i] = vzero();
    gemm<4>(acc, xrow, wih, 256, HD, 0, half, l16);
#pragma unroll
    for (int nt = 0; nt < 8; ++nt) {
        int col = nt * 16 + l16;
#pragma unroll
        for (int v = 0; v < 8; ++v) {
            int row = v + 8 * half;
            S2[wv][row * HD + col] = tanhf(acc[nt][v] + bih[256 + col] + S2[wv][row * HD + col]);
        }
    }
    // z gate + combine
#pragma unroll
    for (int i = 0; i < 8; ++i) acc[i] = vzero();
    gemm<4>(acc, xrow, wih, 128, HD, 0, half, l16);
    gemm<4>(acc, hrow, whh, 128, HD, 0, half, l16);
#pragma unroll
    for (int nt = 0; nt < 8; ++nt) {
        int col = nt * 16 + l16;
#pragma unroll
        for (int v = 0; v < 8; ++v) {
            int row = v + 8 * half;
            int n = n0 + row;
            float z = sigmf(acc[nt][v] + bih[128 + col] + bhh[128 + col]);
            float hv = (n < N) ? hstate[(size_t)n * HD + col] : 0.f;
            float nv = S2[wv][row * HD + col];
            S1[wv][row * HD + col] = nv + z * (hv - nv);
        }
    }
    // row LayerNorm
    for (int r = 0; r < 16; ++r) {
        int n = n0 + r;
        float tv[4], p0 = 0.f;
#pragma unroll
        for (int j = 0; j < 4; ++j) { tv[j] = S1[wv][r * HD + lane + 32 * j]; p0 += tv[j]; }
        float m = wsum(p0) * (1.0f / 128.0f);
        float q = 0.f;
#pragma unroll
        for (int j = 0; j < 4; ++j) { float d = tv[j] - m; q += d * d; }
        float inv = rsqrtf(wsum(q) * (1.0f / 128.0f) + 1e-5f);
        if (n < N) {
#pragma unroll
            for (int j = 0; j < 4; ++j) {
                int c = lane + 32 * j;
                out[(size_t)n * HD + c] = (tv[j] - m) * inv * lng[c] + lnb[c];
            }
        }
    }
}

// ---------------- edge update: eu -> GRU(ge) -> LN ----------------
__global__ __launch_bounds__(128) void k_edge_update(
    const uint16_t* __restrict__ eb16, const uint16_t* __restrict__ msg16,
    const uint16_t* __restrict__ weu, const float* __restrict__ beu,
    const uint16_t* __restrict__ gih, const uint16_t* __restrict__ ghh,
    const float* __restrict__ bih, const float* __restrict__ bhh,
    const float* __restrict__ estate, const uint16_t* __restrict__ es16,
    const float* __restrict__ lng, const float* __restrict__ lnb,
    float* __restrict__ out, int E) {
    __shared__ uint16_t Q[4][16 * HD];
    __shared__ float S1[4][16 * HD];
    __shared__ float S2[4][16 * HD];
    int wv = threadIdx.x >> 5, lane = threadIdx.x & 31;
    int half = lane >> 4, l16 = lane & 15;
    int e0 = (blockIdx.x * 4 + wv) * 16;
    int eM = e0 + l16; if (eM >= E) eM = E - 1;

    v8f acc[8];
#pragma unroll
    for (int i = 0; i < 8; ++i) acc[i] = vzero();
    gemm<4>(acc, eb16 + (size_t)eM * HD, weu, 0, 256, 0, half, l16);
    gemm<4>(acc, msg16 + (size_t)eM * HD, weu, 0, 256, 128, half, l16);
#pragma unroll
    for (int nt = 0; nt < 8; ++nt) {
        int col = nt * 16 + l16;
        float bias = beu[col];
#pragma unroll
        for (int v = 0; v < 8; ++v) {
            int row = v + 8 * half;
            Q[wv][row * HD + col] = f2bf(geluf(acc[nt][v] + bias));
        }
    }
    const uint16_t* hrow = es16 + (size_t)eM * HD;
    const uint16_t* xrow = Q[wv] + (size_t)l16 * HD;

#pragma unroll
    for (int i = 0; i < 8; ++i) acc[i] = vzero();
    gemm<4>(acc, xrow, gih, 0, HD, 0, half, l16);
    gemm<4>(acc, hrow, ghh, 0, HD, 0, half, l16);
#pragma unroll
    for (int nt = 0; nt < 8; ++nt) {
        int col = nt * 16 + l16;
#pragma unroll
        for (int v = 0; v < 8; ++v) {
            int row = v + 8 * half;
            S1[wv][row * HD + col] = sigmf(acc[nt][v] + bih[col] + bhh[col]);
        }
    }
#pragma unroll
    for (int i = 0; i < 8; ++i) acc[i] = vzero();
    gemm<4>(acc, hrow, ghh, 256, HD, 0, half, l16);
#pragma unroll
    for (int nt = 0; nt < 8; ++nt) {
        int col = nt * 16 + l16;
#pragma unroll
        for (int v = 0; v < 8; ++v) {
            int row = v + 8 * half;
            S2[wv][row * HD + col] = S1[wv][row * HD + col] * (acc[nt][v] + bhh[256 + col]);
        }
    }
#pragma unroll
    for (int i = 0; i < 8; ++i) acc[i] = vzero();
    gemm<4>(acc, xrow, gih, 256, HD, 0, half, l16);
#pragma unroll
    for (int nt = 0; nt < 8; ++nt) {
        int col = nt * 16 + l16;
#pragma unroll
        for (int v = 0; v < 8; ++v) {
            int row = v + 8 * half;
            S2[wv][row * HD + col] = tanhf(acc[nt][v] + bih[256 + col] + S2[wv][row * HD + col]);
        }
    }
#pragma unroll
    for (int i = 0; i < 8; ++i) acc[i] = vzero();
    gemm<4>(acc, xrow, gih, 128, HD, 0, half, l16);
    gemm<4>(acc, hrow, ghh, 128, HD, 0, half, l16);
#pragma unroll
    for (int nt = 0; nt < 8; ++nt) {
        int col = nt * 16 + l16;
#pragma unroll
        for (int v = 0; v < 8; ++v) {
            int row = v + 8 * half;
            int e = e0 + row;
            float z = sigmf(acc[nt][v] + bih[128 + col] + bhh[128 + col]);
            float hv = (e < E) ? estate[(size_t)e * HD + col] : 0.f;
            float nv = S2[wv][row * HD + col];
            S1[wv][row * HD + col] = nv + z * (hv - nv);
        }
    }
    for (int r = 0; r < 16; ++r) {
        int e = e0 + r;
        float tv[4], p0 = 0.f;
#pragma unroll
        for (int j = 0; j < 4; ++j) { tv[j] = S1[wv][r * HD + lane + 32 * j]; p0 += tv[j]; }
        float m = wsum(p0) * (1.0f / 128.0f);
        float q = 0.f;
#pragma unroll
        for (int j = 0; j < 4; ++j) { float d = tv[j] - m; q += d * d; }
        float inv = rsqrtf(wsum(q) * (1.0f / 128.0f) + 1e-5f);
        if (e < E) {
#pragma unroll
            for (int j = 0; j < 4; ++j) {
                int c = lane + 32 * j;
                out[(size_t)e * HD + c] = (tv[j] - m) * inv * lng[c] + lnb[c];
            }
        }
    }
}

// ---------------- host launcher ----------------
extern "C" void kernel_launch(void* const* d_in, const int* in_sizes, int n_in,
                              void* d_out, int out_size, void* d_ws, size_t ws_size,
                              hipStream_t stream) {
    (void)n_in; (void)out_size; (void)ws_size;
    const int Hh = 128;
    const int N = in_sizes[0] / Hh;
    const int E = in_sizes[1] / Hh;
    const int NUM1D = 40000;

    const float* node_state = (const float*)d_in[0];
    const float* edge_state = (const float*)d_in[1];
    const float* node_dyn   = (const float*)d_in[2];
    const float* edge_dyn   = (const float*)d_in[3];
    const float* node_stat  = (const float*)d_in[4];
    const float* edge_stat  = (const float*)d_in[5];
    const int*   eidx       = (const int*)d_in[6];
    const int* src = eidx;
    const int* dst = eidx + E;
    auto F = [&](int i) { return (const float*)d_in[i]; };

    // workspace bump allocator
    uintptr_t base = (uintptr_t)d_ws;
    size_t off = 0;
    auto alloc = [&](size_t bytes) -> void* {
        void* p = (void*)(base + off);
        off = (off + bytes + 255) & ~(size_t)255;
        return p;
    };
    uint16_t* wbf   = (uint16_t*)alloc((size_t)524288 * 2);
    float*    nb32  = (float*)alloc((size_t)N * Hh * 4);
    uint16_t* nb16  = (uint16_t*)alloc((size_t)N * Hh * 2);
    uint16_t* ns16  = (uint16_t*)alloc((size_t)N * Hh * 2);
    uint16_t* nh116 = (uint16_t*)alloc((size_t)N * Hh * 2);
    uint16_t* eb16  = (uint16_t*)alloc((size_t)E * Hh * 2);
    uint16_t* es16  = (uint16_t*)alloc((size_t)E * Hh * 2);
    uint16_t* msg16 = (uint16_t*)alloc((size_t)E * Hh * 2);
    float*    agg   = (float*)alloc((size_t)N * Hh * 4);
    float*    agg2  = (float*)alloc((size_t)N * Hh * 4);
    float*    indeg = (float*)alloc((size_t)N * 4);

    // bf16 weight table: m1a m1b mn1 m2a m2b mn2 nu eu g1ih g1hh g2ih g2hh geih gehh
    static const unsigned offs[14] = {0, 49152, 65536, 81920, 131072, 147456, 163840,
                                      196608, 229376, 278528, 327680, 376832, 425984, 475136};
    static const unsigned cnts[14] = {49152, 16384, 16384, 49152, 16384, 16384, 32768,
                                      32768, 49152, 49152, 49152, 49152, 49152, 49152};
    static const int widx[14] = {15, 17, 19, 21, 23, 25, 27, 29, 31, 32, 35, 36, 39, 40};
    WTab t;
    for (int i = 0; i < 14; ++i) { t.w[i].src = F(widx[i]); t.w[i].dstOff = offs[i]; t.w[i].cnt = cnts[i]; }
    k_conv<<<256, 256, 0, stream>>>(t, wbf);

    k_encode_nodes<<<(N + 7) / 8, 256, 0, stream>>>(
        node_dyn, node_stat, node_state, F(7), F(8), F(9), F(10),
        nb32, nb16, ns16, agg, agg2, indeg, N);
    k_encode_edges<<<(E + 7) / 8, 256, 0, stream>>>(
        edge_dyn, edge_stat, edge_state, F(11), F(12), F(13), F(14),
        dst, eb16, es16, indeg, E);

    k_edge_msg<1><<<(E + 127) / 128, 256, 0, stream>>>(
        nb16, eb16, src, dst, wbf + offs[0], F(16), wbf + offs[1], F(18), agg, msg16, E);
    k_node_hop1<<<(N + 127) / 128, 256, 0, stream>>>(
        agg, indeg, wbf + offs[2], F(20), nb32, nh116, N);
    k_edge_msg<2><<<(E + 127) / 128, 256, 0, stream>>>(
        nh116, eb16, src, dst, wbf + offs[3], F(22), wbf + offs[4], F(24), agg2, nullptr, E);

    float* out = (float*)d_out;
    k_node_update<<<(N + 63) / 64, 128, 0, stream>>>(
        agg2, indeg, wbf + offs[5], F(26), nh116, wbf + offs[6], F(28),
        wbf + offs[8], wbf + offs[9], F(33), F(34),
        wbf + offs[10], wbf + offs[11], F(37), F(38),
        node_state, ns16, F(43), F(44), out, N, NUM1D);
    k_edge_update<<<(E + 63) / 64, 128, 0, stream>>>(
        eb16, msg16, wbf + offs[7], F(30),
        wbf + offs[12], wbf + offs[13], F(41), F(42),
        edge_state, es16, F(45), F(46), out + (size_t)N * Hh, E);
}
